// HybridBlock_37237366456738
// MI455X (gfx1250) — matrix operand order repaired and checked by reading, not compiled, not run on hardware
//
#include <hip/hip_runtime.h>
#include <hip/hip_bf16.h>
#include <math.h>

// ---------------------------------------------------------------------------
// CDNA5 (gfx1250) wave32 WMMA implementation of the hybrid transformer block.
// ---------------------------------------------------------------------------

typedef __attribute__((ext_vector_type(16))) __bf16 v16bf;
typedef __attribute__((ext_vector_type(8)))  __bf16 v8bf;
typedef __attribute__((ext_vector_type(8)))  float  v8f;
typedef int i4n __attribute__((__vector_size__(16)));   // <4 x i32> for async-LDS builtin

#define LL   2048
#define DD   2048
#define HH   32
#define HKVC 8
#define HDC  64
#define EE   8
#define DFFC 8192
#define KVD  512

// Async global->LDS staging (ASYNCcnt path), if the toolchain exposes it.
#if defined(__AMDGCN__) && __has_builtin(__builtin_amdgcn_global_load_async_to_lds_b128) && __has_builtin(__builtin_amdgcn_s_wait_asynccnt)
#define ASYNC_LDS 1
#else
#define ASYNC_LDS 0
#endif

// ===========================================================================
// Generic bf16 WMMA GEMM:  C[M,N] (+opts) = act(A[M,K]_bf16 @ B[K,N]_f32)
//   ACT:    0 none, 1 exact GELU
//   OUTBF:  0 f32 out, 1 bf16 out
//   MODE:   0 store, 1 out = Res + v, 2 out += v
//   SCALED: per-row scale RS[row*rsStride]
// Block tile 128x128, K-step 32; 8 waves, each wave 32(M)x64(N).
// ===========================================================================
template<int ACT, int OUTBF, int MODE, int SCALED>
__global__ __launch_bounds__(256) void gemm_bf16_wmma(
    const __bf16* __restrict__ A, const float* __restrict__ B,
    void* __restrict__ C, const float* __restrict__ Res,
    const float* __restrict__ RS, int M, int N, int Kd, int rsStride)
{
  __shared__ alignas(32) __bf16 As[128][48];   // [m][k], padded
  __shared__ alignas(32) __bf16 Bs[128][48];   // [n][k] transposed, padded

  const int tid  = threadIdx.x;
  const int wid  = tid >> 5, lane = tid & 31;
  const int half = lane >> 4, l16 = lane & 15;
  const int m0 = blockIdx.y * 128, n0 = blockIdx.x * 128;
  const int wm = (wid >> 1) * 32,  wn = (wid & 1) * 64;

  v8f acc[2][4];
#pragma unroll
  for (int a = 0; a < 2; ++a)
#pragma unroll
    for (int b = 0; b < 4; ++b)
#pragma unroll
      for (int i = 0; i < 8; ++i) acc[a][b][i] = 0.f;

  const int aRow = tid >> 1, aK = (tid & 1) * 16;    // A: 16 bf16 per thread
  const int bK   = tid >> 3, bN = (tid & 7) * 16;    // B: 16 f32 per thread

  for (int k0 = 0; k0 < Kd; k0 += 32) {
    // ---- stage A tile (already bf16): async DMA to LDS when available ----
    const __bf16* agp = A + (size_t)(m0 + aRow) * Kd + k0 + aK;
#if ASYNC_LDS
    __builtin_amdgcn_global_load_async_to_lds_b128(
        (i4n __attribute__((address_space(1)))*)agp,
        (i4n __attribute__((address_space(3)))*)&As[aRow][aK], 0, 0);
    __builtin_amdgcn_global_load_async_to_lds_b128(
        (i4n __attribute__((address_space(1)))*)(agp + 8),
        (i4n __attribute__((address_space(3)))*)&As[aRow][aK + 8], 0, 0);
#else
    {
      const uint4* asrc = reinterpret_cast<const uint4*>(agp);
      uint4 a0 = asrc[0], a1 = asrc[1];
      *reinterpret_cast<uint4*>(&As[aRow][aK])     = a0;
      *reinterpret_cast<uint4*>(&As[aRow][aK + 8]) = a1;
    }
#endif
    // ---- stage B tile (f32 -> bf16, transposed) ----
    const float4* bsrc =
        reinterpret_cast<const float4*>(B + (size_t)(k0 + bK) * N + n0 + bN);
#pragma unroll
    for (int j = 0; j < 4; ++j) {
      float4 f = bsrc[j];
      Bs[bN + j * 4 + 0][bK] = (__bf16)f.x;
      Bs[bN + j * 4 + 1][bK] = (__bf16)f.y;
      Bs[bN + j * 4 + 2][bK] = (__bf16)f.z;
      Bs[bN + j * 4 + 3][bK] = (__bf16)f.w;
    }
    if (k0 + 32 < Kd) {  // prefetch next tiles (global_prefetch_b8)
      __builtin_prefetch(A + (size_t)(m0 + aRow) * Kd + k0 + 32 + aK, 0, 0);
      __builtin_prefetch(B + (size_t)(k0 + 32 + bK) * N + n0 + bN, 0, 0);
    }
#if ASYNC_LDS
    __builtin_amdgcn_s_wait_asynccnt(0);   // A-tile DMA visible in LDS
#endif
    __syncthreads();

    // ---- per-lane fragments (contiguous k = 16*half + s, same order in A and B) ----
    v16bf af[2], bfv[4];
#pragma unroll
    for (int im = 0; im < 2; ++im) {
      const int m = wm + im * 16 + l16;
      v8bf lo = *reinterpret_cast<const v8bf*>(&As[m][16 * half]);
      v8bf hi = *reinterpret_cast<const v8bf*>(&As[m][16 * half + 8]);
#pragma unroll
      for (int i = 0; i < 8; ++i) { af[im][i] = lo[i]; af[im][i + 8] = hi[i]; }
    }
#pragma unroll
    for (int in = 0; in < 4; ++in) {
      const int n = wn + in * 16 + l16;
      bfv[in] = *reinterpret_cast<const v16bf*>(&Bs[n][16 * half]);
    }
#pragma unroll
    for (int im = 0; im < 2; ++im)
#pragma unroll
      for (int in = 0; in < 4; ++in)
        acc[im][in] = __builtin_amdgcn_wmma_f32_16x16x32_bf16(
            false, af[im], false, bfv[in], (short)0, acc[im][in], false, false);
    __syncthreads();
  }

  // ---- epilogue (C layout: VGPR r -> row r + 8*half, col = l16) ----
#pragma unroll
  for (int im = 0; im < 2; ++im) {
#pragma unroll
    for (int in = 0; in < 4; ++in) {
#pragma unroll
      for (int r = 0; r < 8; ++r) {
        const int row = m0 + wm + im * 16 + r + 8 * half;
        const int col = n0 + wn + in * 16 + l16;
        const size_t idx = (size_t)row * N + col;
        float v = acc[im][in][r];
        if (ACT == 1) v = 0.5f * v * (1.f + erff(v * 0.70710678118654752f));
        if (SCALED)   v *= RS[(size_t)row * rsStride];
        if (MODE == 1) v += Res[idx];
        if (MODE == 2) v += reinterpret_cast<float*>(C)[idx];
        if (OUTBF) reinterpret_cast<__bf16*>(C)[idx] = (__bf16)v;
        else       reinterpret_cast<float*>(C)[idx]  = v;
      }
    }
  }
}

// ===========================================================================
// RMSNorm (fp32 in, bf16 out), one block per token row.
// ===========================================================================
__global__ __launch_bounds__(256) void rmsnorm_bf16(
    const float* __restrict__ x, const float* __restrict__ w,
    __bf16* __restrict__ out)
{
  const int row = blockIdx.x, tid = threadIdx.x;
  const int wid = tid >> 5, lane = tid & 31;
  const float* xr = x + (size_t)row * DD;
  float ss = 0.f;
  for (int i = tid; i < DD; i += 256) { float t = xr[i]; ss += t * t; }
#pragma unroll
  for (int m = 16; m >= 1; m >>= 1) ss += __shfl_xor(ss, m, 32);
  __shared__ float red[8];
  if (lane == 0) red[wid] = ss;
  __syncthreads();
  if (tid == 0) {
    float t = 0.f;
#pragma unroll
    for (int i = 0; i < 8; ++i) t += red[i];
    red[0] = rsqrtf(t / (float)DD + 1e-6f);
  }
  __syncthreads();
  const float inv = red[0];
  for (int i = tid; i < DD; i += 256)
    out[(size_t)row * DD + i] = (__bf16)(xr[i] * inv * w[i]);
}

// ===========================================================================
// RoPE + repack [L][nh][64] f32 -> [nh][L][64] bf16   (hshift = log2(nh))
// ===========================================================================
__global__ __launch_bounds__(256) void rope_pack(
    const float* __restrict__ src, __bf16* __restrict__ dst, int hshift)
{
  const int idx = blockIdx.x * 256 + threadIdx.x;
  const int i = idx & 63;
  const int h = (idx >> 6) & ((1 << hshift) - 1);
  const int t = idx >> (6 + hshift);
  const int j = i & 31;
  const float ang = (float)t * __expf((float)j * (-2.f / 64.f) * 9.210340371976184f);
  float sn, cs;
  __sincosf(ang, &sn, &cs);
  const float xv  = src[idx];
  const float rot = (i < 32) ? -src[idx + 32] : src[idx - 32];
  dst[((size_t)h * LL + t) * 64 + i] = (__bf16)(xv * cs + rot * sn);
}

// V: [L][Hkv][64] f32 -> [Hkv][64][L] bf16  (so PV B-frags are K-contiguous)
__global__ __launch_bounds__(256) void v_transpose_cast(
    const float* __restrict__ v, __bf16* __restrict__ vt)
{
  const int idx = blockIdx.x * 256 + threadIdx.x;  // over L*512
  const int i = idx & 63;
  const int hk = (idx >> 6) & 7;
  const int t = idx >> 9;
  vt[((size_t)hk * 64 + i) * LL + t] = (__bf16)v[idx];
}

// ===========================================================================
// Flash attention, GQA (4 Q heads per KV head), causal.
// grid = (L/128, H); 8 waves/block, each wave owns 16 q rows of one head.
// QK^T and P@V both on v_wmma_f32_16x16x32_bf16.
// ===========================================================================
__global__ __launch_bounds__(256) void attn_flash(
    const __bf16* __restrict__ qb,   // [H][L][64]
    const __bf16* __restrict__ kb,   // [Hkv][L][64]
    const __bf16* __restrict__ vt,   // [Hkv][64][L]
    __bf16* __restrict__ ctx)        // [L][H*64] bf16
{
  __shared__ alignas(32) __bf16 Ps[8][16][64];   // per-wave P tile
  const int h = blockIdx.y, hk = h >> 2;
  const int tid = threadIdx.x, wid = tid >> 5, lane = tid & 31;
  const int half = lane >> 4, l16 = lane & 15;
  const int qbase = blockIdx.x * 128 + wid * 16;

  // Q fragments (A layout), loaded once
  v16bf qf[2];
  {
    const __bf16* qr = qb + ((size_t)h * LL + qbase + l16) * 64;
#pragma unroll
    for (int kk = 0; kk < 2; ++kk) {
      v8bf lo = *reinterpret_cast<const v8bf*>(qr + kk * 32 + 16 * half);
      v8bf hi = *reinterpret_cast<const v8bf*>(qr + kk * 32 + 16 * half + 8);
#pragma unroll
      for (int i = 0; i < 8; ++i) { qf[kk][i] = lo[i]; qf[kk][i + 8] = hi[i]; }
    }
  }

  v8f o[4];
#pragma unroll
  for (int nt = 0; nt < 4; ++nt)
#pragma unroll
    for (int r = 0; r < 8; ++r) o[nt][r] = 0.f;
  float mst[8], lst[8];
#pragma unroll
  for (int r = 0; r < 8; ++r) { mst[r] = -3.0e38f; lst[r] = 0.f; }

  const int kend = (qbase + 15) >> 6;              // last 64-wide K tile
  for (int kt = 0; kt <= kend; ++kt) {
    const int kbase = kt * 64;
    // ---- S = Q K^T (16x64), K fragments straight from global ----
    v8f s[4];
#pragma unroll
    for (int nt = 0; nt < 4; ++nt) {
#pragma unroll
      for (int r = 0; r < 8; ++r) s[nt][r] = 0.f;
      const __bf16* kr = kb + ((size_t)hk * LL + kbase + nt * 16 + l16) * 64;
#pragma unroll
      for (int kk = 0; kk < 2; ++kk) {
        v16bf bfr = *reinterpret_cast<const v16bf*>(kr + kk * 32 + 16 * half);
        s[nt] = __builtin_amdgcn_wmma_f32_16x16x32_bf16(
            false, qf[kk], false, bfr, (short)0, s[nt], false, false);
      }
    }
    // ---- scale + causal mask (only the diagonal tile can mask) ----
    const bool doMask = (kt == kend);
#pragma unroll
    for (int nt = 0; nt < 4; ++nt) {
      const int col = kbase + nt * 16 + l16;
#pragma unroll
      for (int r = 0; r < 8; ++r) {
        const int row = qbase + r + 8 * half;
        float v = s[nt][r] * 0.125f;                 // 1/sqrt(64)
        s[nt][r] = (doMask && col > row) ? -1.0e9f : v;
      }
    }
    // ---- online softmax (rows live in 16-lane halves) ----
#pragma unroll
    for (int r = 0; r < 8; ++r) {
      float mt = fmaxf(fmaxf(s[0][r], s[1][r]), fmaxf(s[2][r], s[3][r]));
#pragma unroll
      for (int msk = 8; msk >= 1; msk >>= 1) mt = fmaxf(mt, __shfl_xor(mt, msk, 32));
      const float mn    = fmaxf(mst[r], mt);
      const float alpha = __expf(mst[r] - mn);
      float rs = 0.f;
#pragma unroll
      for (int nt = 0; nt < 4; ++nt) {
        float p = __expf(s[nt][r] - mn);
        s[nt][r] = p;
        rs += p;
      }
#pragma unroll
      for (int msk = 8; msk >= 1; msk >>= 1) rs += __shfl_xor(rs, msk, 32);
      lst[r] = lst[r] * alpha + rs;
      mst[r] = mn;
#pragma unroll
      for (int nt = 0; nt < 4; ++nt) o[nt][r] *= alpha;
    }
    // ---- C-layout -> A-layout via per-wave LDS ----
#pragma unroll
    for (int nt = 0; nt < 4; ++nt)
#pragma unroll
      for (int r = 0; r < 8; ++r)
        Ps[wid][r + 8 * half][nt * 16 + l16] = (__bf16)s[nt][r];
    asm volatile("s_wait_dscnt 0" ::: "memory");
    // ---- O += P @ V ----
#pragma unroll
    for (int kt2 = 0; kt2 < 2; ++kt2) {
      v16bf pa;
      v8bf lo = *reinterpret_cast<const v8bf*>(&Ps[wid][l16][kt2 * 32 + 16 * half]);
      v8bf hi = *reinterpret_cast<const v8bf*>(&Ps[wid][l16][kt2 * 32 + 16 * half + 8]);
#pragma unroll
      for (int i = 0; i < 8; ++i) { pa[i] = lo[i]; pa[i + 8] = hi[i]; }
#pragma unroll
      for (int nt = 0; nt < 4; ++nt) {
        const __bf16* vr = vt + ((size_t)hk * 64 + nt * 16 + l16) * LL
                              + kbase + kt2 * 32 + 16 * half;
        v16bf bv = *reinterpret_cast<const v16bf*>(vr);
        o[nt] = __builtin_amdgcn_wmma_f32_16x16x32_bf16(
            false, pa, false, bv, (short)0, o[nt], false, false);
      }
    }
  }
  // ---- normalize, store ctx (bf16 for the Wo GEMM) ----
#pragma unroll
  for (int nt = 0; nt < 4; ++nt)
#pragma unroll
    for (int r = 0; r < 8; ++r) {
      const int row = qbase + r + 8 * half;
      const int col = h * 64 + nt * 16 + l16;
      ctx[(size_t)row * DD + col] = (__bf16)(o[nt][r] / lst[r]);
    }
}

// ===========================================================================
// Gate: logits = h2 @ gate_w [D,8]; top-2 softmax -> dense combine weights.
// One wave per token.
// ===========================================================================
__global__ __launch_bounds__(256) void gate_topk(
    const __bf16* __restrict__ h2, const float* __restrict__ gw,
    float* __restrict__ comb)
{
  const int wid = threadIdx.x >> 5, lane = threadIdx.x & 31;
  const int t = blockIdx.x * 8 + wid;
  const __bf16* hr = h2 + (size_t)t * DD;
  float acc[8];
#pragma unroll
  for (int e = 0; e < 8; ++e) acc[e] = 0.f;
  for (int i = lane; i < DD; i += 32) {
    const float hv = (float)hr[i];
    const float* g = gw + (size_t)i * 8;
#pragma unroll
    for (int e = 0; e < 8; ++e) acc[e] += hv * g[e];
  }
#pragma unroll
  for (int e = 0; e < 8; ++e)
#pragma unroll
    for (int m = 16; m >= 1; m >>= 1) acc[e] += __shfl_xor(acc[e], m, 32);
  if (lane == 0) {
    int i0 = 0; float v0 = acc[0];
    for (int e = 1; e < 8; ++e) if (acc[e] > v0) { v0 = acc[e]; i0 = e; }
    int i1 = -1; float v1 = -3.0e38f;
    for (int e = 0; e < 8; ++e)
      if (e != i0 && acc[e] > v1) { v1 = acc[e]; i1 = e; }
    const float e1 = __expf(v1 - v0);
    const float g0 = 1.f / (1.f + e1), g1 = e1 / (1.f + e1);
#pragma unroll
    for (int e = 0; e < 8; ++e)
      comb[(size_t)t * 8 + e] = (e == i0) ? g0 : ((e == i1) ? g1 : 0.f);
  }
}

// ===========================================================================
// Host-side orchestration.
// Inputs: 0:x 1:mask 2:attn_norm_w 3:ffn_norm_w 4:Wq 5:Wk 6:Wv 7:Wo
//         8:gate_w 9:W1 10:W2        (mask is reproduced analytically)
// ===========================================================================
extern "C" void kernel_launch(void* const* d_in, const int* in_sizes, int n_in,
                              void* d_out, int out_size, void* d_ws, size_t ws_size,
                              hipStream_t stream)
{
  const float* x   = (const float*)d_in[0];
  const float* anw = (const float*)d_in[2];
  const float* fnw = (const float*)d_in[3];
  const float* Wq  = (const float*)d_in[4];
  const float* Wk  = (const float*)d_in[5];
  const float* Wv  = (const float*)d_in[6];
  const float* Wo  = (const float*)d_in[7];
  const float* gw  = (const float*)d_in[8];
  const float* W1  = (const float*)d_in[9];
  const float* W2  = (const float*)d_in[10];
  float* out = (float*)d_out;

  char* p = (char*)d_ws;
  auto alloc = [&](size_t bytes) {
    char* r = p;
    p += (bytes + 255) & ~(size_t)255;
    return r;
  };
  __bf16* h_b   = (__bf16*)alloc((size_t)LL * DD * 2);
  __bf16* h2_b  = (__bf16*)alloc((size_t)LL * DD * 2);
  float*  q_f   = (float*) alloc((size_t)LL * DD * 4);
  float*  k_f   = (float*) alloc((size_t)LL * KVD * 4);
  float*  v_f   = (float*) alloc((size_t)LL * KVD * 4);
  __bf16* qb    = (__bf16*)alloc((size_t)HH * LL * 64 * 2);
  __bf16* kb    = (__bf16*)alloc((size_t)HKVC * LL * 64 * 2);
  __bf16* vt    = (__bf16*)alloc((size_t)HKVC * 64 * LL * 2);
  __bf16* ctx_b = (__bf16*)alloc((size_t)LL * DD * 2);
  float*  x2    = (float*) alloc((size_t)LL * DD * 4);
  __bf16* hid_b = (__bf16*)alloc((size_t)LL * DFFC * 2);
  float*  comb  = (float*) alloc((size_t)LL * EE * 4);

  const dim3 blk(256);

  // ---- attention branch ----
  rmsnorm_bf16<<<LL, blk, 0, stream>>>(x, anw, h_b);
  gemm_bf16_wmma<0,0,0,0><<<dim3(DD / 128, LL / 128), blk, 0, stream>>>(
      h_b, Wq, q_f, nullptr, nullptr, LL, DD, DD, 0);
  gemm_bf16_wmma<0,0,0,0><<<dim3(KVD / 128, LL / 128), blk, 0, stream>>>(
      h_b, Wk, k_f, nullptr, nullptr, LL, KVD, DD, 0);
  gemm_bf16_wmma<0,0,0,0><<<dim3(KVD / 128, LL / 128), blk, 0, stream>>>(
      h_b, Wv, v_f, nullptr, nullptr, LL, KVD, DD, 0);
  rope_pack<<<(LL * DD) / 256, blk, 0, stream>>>(q_f, qb, 5);
  rope_pack<<<(LL * KVD) / 256, blk, 0, stream>>>(k_f, kb, 3);
  v_transpose_cast<<<(LL * KVD) / 256, blk, 0, stream>>>(v_f, vt);
  attn_flash<<<dim3(LL / 128, HH), blk, 0, stream>>>(qb, kb, vt, ctx_b);
  gemm_bf16_wmma<0,0,1,0><<<dim3(DD / 128, LL / 128), blk, 0, stream>>>(
      ctx_b, Wo, x2, x, nullptr, LL, DD, DD, 0);

  // ---- MoE branch ----
  rmsnorm_bf16<<<LL, blk, 0, stream>>>(x2, fnw, h2_b);
  gate_topk<<<LL / 8, blk, 0, stream>>>(h2_b, gw, comb);
  for (int e = 0; e < EE; ++e) {
    const float* W1e = W1 + (size_t)e * DD * DFFC;
    const float* W2e = W2 + (size_t)e * DFFC * DD;
    gemm_bf16_wmma<1,1,0,0><<<dim3(DFFC / 128, LL / 128), blk, 0, stream>>>(
        h2_b, W1e, hid_b, nullptr, nullptr, LL, DFFC, DD, 0);
    if (e == 0)
      gemm_bf16_wmma<0,0,1,1><<<dim3(DD / 128, LL / 128), blk, 0, stream>>>(
          hid_b, W2e, out, x2, comb + e, LL, DD, DFFC, EE);
    else
      gemm_bf16_wmma<0,0,2,1><<<dim3(DD / 128, LL / 128), blk, 0, stream>>>(
          hid_b, W2e, out, nullptr, comb + e, LL, DD, DFFC, EE);
  }
  (void)in_sizes; (void)n_in; (void)out_size; (void)ws_size; (void)p;
}
